// ODE2VAE_77180562309503
// MI455X (gfx1250) — matrix operand order repaired and checked by reading, not compiled, not run on hardware
//
#include <hip/hip_runtime.h>
#include <hip/hip_bf16.h>

#define N_B 512
#define T_N 16
#define SD 17
#define LOG2PI 1.8378770664093453f
#define LDP 40   // LDS row pitch (elements); 80B rows keep 16B fragment alignment

typedef __bf16 bf16_t;
typedef __attribute__((ext_vector_type(16))) __bf16 v16bf;
typedef __attribute__((ext_vector_type(8))) float v8f;

__device__ __forceinline__ unsigned short f2bf_u(float f) {
  union { float f; unsigned u; } c; c.f = f;
  unsigned r = c.u + 0x7fffu + ((c.u >> 16) & 1u);
  return (unsigned short)(r >> 16);
}
__device__ __forceinline__ unsigned pack2bf(float a, float b) {
  return (unsigned)f2bf_u(a) | ((unsigned)f2bf_u(b) << 16);
}

__device__ __forceinline__ float wave_sum(float v) {
#pragma unroll
  for (int off = 16; off >= 1; off >>= 1) v += __shfl_xor(v, off, 32);
  return v;
}

__device__ __forceinline__ float block_reduce_sum(float v) {
  __shared__ float sm[8];
  v = wave_sum(v);
  int lane = threadIdx.x & 31, w = threadIdx.x >> 5;
  if (lane == 0) sm[w] = v;
  __syncthreads();
  int nw = (blockDim.x + 31) >> 5;
  v = (threadIdx.x < (unsigned)nw) ? sm[threadIdx.x] : 0.f;
  if (w == 0) v = wave_sum(v);
  return v;
}

__global__ void zero_kernel(float* p, int n) {
  int i = blockIdx.x * blockDim.x + threadIdx.x;
  if (i < n) p[i] = 0.f;
}

// ---------------- encoder: direct conv + relu (NCHW, small) ----------------
__global__ void conv_relu_kernel(const float* __restrict__ x, const float* __restrict__ w,
                                 const float* __restrict__ b, float* __restrict__ y,
                                 int N, int Cin, int Hin, int Win, int Cout, int Hout, int Wout,
                                 int Kk, int stride, int pad, int xImgStride) {
  int o = blockIdx.x * blockDim.x + threadIdx.x;
  int total = N * Cout * Hout * Wout;
  if (o >= total) return;
  int ox = o % Wout; int t = o / Wout;
  int oy = t % Hout; t /= Hout;
  int co = t % Cout; int n = t / Cout;
  float acc = b[co];
  const float* xb = x + (size_t)n * xImgStride;
  for (int ky = 0; ky < Kk; ++ky) {
    int iy = oy * stride - pad + ky; if (iy < 0 || iy >= Hin) continue;
    for (int kx = 0; kx < Kk; ++kx) {
      int ix = ox * stride - pad + kx; if (ix < 0 || ix >= Win) continue;
      const float* wp = w + ((size_t)(ky * Kk + kx) * Cin) * Cout + co;
      const float* xp = xb + (size_t)iy * Win + ix;
      for (int ci = 0; ci < Cin; ++ci)
        acc += xp[(size_t)ci * Hin * Win] * wp[(size_t)ci * Cout];
    }
  }
  y[o] = fmaxf(acc, 0.f);
}

// ---------------- small FC ----------------
__global__ void fc_kernel(const float* __restrict__ x, const float* __restrict__ w,
                          const float* __restrict__ b, float* __restrict__ y,
                          int N, int K, int M) {
  int e = blockIdx.x * blockDim.x + threadIdx.x;
  if (e >= N * M) return;
  int n = e / M, j = e - n * M;
  float acc = b[j];
  const float* xr = x + (size_t)n * K;
  for (int k = 0; k < K; ++k) acc += xr[k] * w[(size_t)k * M + j];
  y[e] = acc;
}

__global__ void init_y0_kernel(const float* __restrict__ qm, const float* __restrict__ qv,
                               const float* __restrict__ eps, float* __restrict__ y0) {
  int n = blockIdx.x * blockDim.x + threadIdx.x;
  if (n >= N_B) return;
  float ss = 0.f;
  for (int d = 0; d < 16; ++d) {
    float e = eps[n * 16 + d];
    ss += e * e;
    y0[n * SD + d] = qm[n * 16 + d] + e * expf(qv[n * 16 + d]);
  }
  y0[n * SD + 16] = -0.5f * (ss + 16.f * LOG2PI);
}

// ---------------- BNN forward (one wave per row) ----------------
__global__ __launch_bounds__(32)
void bnn_forward_kernel(const float* __restrict__ y,
                        const float* __restrict__ W1, const float* __restrict__ B1,
                        const float* __restrict__ W2, const float* __restrict__ B2,
                        const float* __restrict__ W3, const float* __restrict__ B3,
                        float* __restrict__ l1g, float* __restrict__ dc1g, float* __restrict__ r1g,
                        float* __restrict__ l2g, float* __restrict__ dc2g, float* __restrict__ r2g,
                        float* __restrict__ dvg) {
  __shared__ float vs[16];
  __shared__ float hb[50];
  const int n = blockIdx.x;
  const int lane = threadIdx.x;
  const float* yr = y + n * SD;
  if (lane < 16) vs[lane] = yr[lane];
  __syncthreads();
  const int j1 = lane, j2 = lane + 32;
  const bool v2 = (j2 < 50);
  float a1 = B1[j1];
  float a2 = v2 ? B1[j2] : 0.f;
  for (int k = 0; k < 16; ++k) {
    float xv = vs[k];
    a1 += xv * W1[k * 50 + j1];
    if (v2) a2 += xv * W1[k * 50 + j2];
  }
  float mean = wave_sum(a1 + (v2 ? a2 : 0.f)) * 0.02f;
  float d1 = a1 - mean, d2 = v2 ? (a2 - mean) : 0.f;
  float var = wave_sum(d1 * d1 + d2 * d2) * 0.02f;
  float r = rsqrtf(var + 1e-5f);
  float l1a = d1 * r, l1b = d2 * r;
  float h1a = l1a > 0.f ? l1a : expm1f(l1a);
  float h1b = l1b > 0.f ? l1b : expm1f(l1b);
  l1g[n * 50 + j1] = l1a;
  dc1g[n * 50 + j1] = l1a > 0.f ? 1.f : (h1a + 1.f);
  hb[j1] = h1a;
  if (v2) { l1g[n * 50 + j2] = l1b; dc1g[n * 50 + j2] = l1b > 0.f ? 1.f : (h1b + 1.f); hb[j2] = h1b; }
  if (lane == 0) r1g[n] = r;
  __syncthreads();
  float c1 = B2[j1];
  float c2 = v2 ? B2[j2] : 0.f;
  for (int k = 0; k < 50; ++k) {
    float xv = hb[k];
    c1 += xv * W2[k * 50 + j1];
    if (v2) c2 += xv * W2[k * 50 + j2];
  }
  mean = wave_sum(c1 + (v2 ? c2 : 0.f)) * 0.02f;
  d1 = c1 - mean; d2 = v2 ? (c2 - mean) : 0.f;
  var = wave_sum(d1 * d1 + d2 * d2) * 0.02f;
  r = rsqrtf(var + 1e-5f);
  float l2a = d1 * r, l2b = d2 * r;
  float h2a = l2a > 0.f ? l2a : expm1f(l2a);
  float h2b = l2b > 0.f ? l2b : expm1f(l2b);
  l2g[n * 50 + j1] = l2a;
  dc2g[n * 50 + j1] = l2a > 0.f ? 1.f : (h2a + 1.f);
  if (v2) { l2g[n * 50 + j2] = l2b; dc2g[n * 50 + j2] = l2b > 0.f ? 1.f : (h2b + 1.f); }
  if (lane == 0) r2g[n] = r;
  __syncthreads();
  hb[j1] = h2a;
  if (v2) hb[j2] = h2b;
  __syncthreads();
  if (lane < 8) {
    float o = B3[lane];
    for (int k = 0; k < 50; ++k) o += hb[k] * W3[k * 8 + lane];
    dvg[n * 8 + lane] = o;
  }
}

// ---------------- BNN JVP trace element (one wave per (dir i, row n)) ----------------
__global__ __launch_bounds__(32)
void bnn_tangent_kernel(const float* __restrict__ W1, const float* __restrict__ W2,
                        const float* __restrict__ W3,
                        const float* __restrict__ l1g, const float* __restrict__ dc1g,
                        const float* __restrict__ r1g,
                        const float* __restrict__ l2g, const float* __restrict__ dc2g,
                        const float* __restrict__ r2g, float* __restrict__ tout) {
  __shared__ float th1[50];
  const int id = blockIdx.x;
  const int n = id >> 3;
  const int i = id & 7;
  const int lane = threadIdx.x;
  const int j1 = lane, j2 = lane + 32;
  const bool v2 = (j2 < 50);
  const float* l1r = l1g + n * 50;
  const float* dc1r = dc1g + n * 50;
  const float* l2r = l2g + n * 50;
  const float* dc2r = dc2g + n * 50;
  const float r1 = r1g[n], r2 = r2g[n];
  float w1a = W1[i * 50 + j1];
  float w1b = v2 ? W1[i * 50 + j2] : 0.f;
  float l1a = l1r[j1];
  float l1b = v2 ? l1r[j2] : 0.f;
  float mw = wave_sum(w1a + w1b) * 0.02f;
  float cc = wave_sum(l1a * w1a + l1b * w1b) * 0.02f;
  float tl1a = r1 * (w1a - mw - l1a * cc);
  float tl1b = r1 * (w1b - mw - l1b * cc);
  th1[j1] = dc1r[j1] * tl1a;
  if (v2) th1[j2] = dc1r[j2] * tl1b;
  __syncthreads();
  float ta = 0.f, tb = 0.f;
  for (int k = 0; k < 50; ++k) {
    float t = th1[k];
    ta += t * W2[k * 50 + j1];
    if (v2) tb += t * W2[k * 50 + j2];
  }
  if (!v2) tb = 0.f;
  float l2a = l2r[j1];
  float l2b = v2 ? l2r[j2] : 0.f;
  float m2 = wave_sum(ta + tb) * 0.02f;
  float c2 = wave_sum(l2a * ta + l2b * tb) * 0.02f;
  float th2a = dc2r[j1] * r2 * (ta - m2 - l2a * c2);
  float th2b = v2 ? (dc2r[j2] * r2 * (tb - m2 - l2b * c2)) : 0.f;
  float contrib = th2a * W3[j1 * 8 + i] + (v2 ? th2b * W3[j2 * 8 + i] : 0.f);
  float tr = wave_sum(contrib);
  if (lane == 0) tout[i * N_B + n] = tr;
}

__global__ void assemble_k_kernel(const float* __restrict__ ys, const float* __restrict__ dv,
                                  const float* __restrict__ tout, float* __restrict__ kout) {
  int n = blockIdx.x * blockDim.x + threadIdx.x;
  if (n >= N_B) return;
#pragma unroll
  for (int d = 0; d < 8; ++d) kout[n * SD + d] = dv[n * 8 + d];
#pragma unroll
  for (int d = 0; d < 8; ++d) kout[n * SD + 8 + d] = ys[n * SD + d];
  float tr = 0.f;
#pragma unroll
  for (int i = 0; i < 8; ++i) tr += tout[i * N_B + n];
  kout[n * SD + 16] = -tr;
}

__global__ void ode_combine_kernel(const float* __restrict__ y,
                                   const float* __restrict__ k1, const float* __restrict__ k2,
                                   const float* __restrict__ k3, const float* __restrict__ k4,
                                   const float* __restrict__ k5, const float* __restrict__ k6,
                                   float c1, float c2, float c3, float c4, float c5, float c6,
                                   float h, float* __restrict__ out) {
  int i = blockIdx.x * blockDim.x + threadIdx.x;
  if (i >= N_B * SD) return;
  float s = c1 * k1[i] + c2 * k2[i] + c3 * k3[i] + c4 * k4[i] + c5 * k5[i] + c6 * k6[i];
  out[i] = y[i] + h * s;
}

__global__ void ztl_klz_kernel(const float* __restrict__ traj, float* __restrict__ ztl,
                               float* __restrict__ st, float* __restrict__ klz_acc) {
  int e = blockIdx.x * blockDim.x + threadIdx.x;
  float c = 0.f;
  if (e < N_B * T_N) {
    int n = e >> 4, t = e & 15;
    const float* row = traj + ((size_t)t * N_B + n) * SD;
    float ss = 0.f;
    for (int d = 0; d < 16; ++d) {
      float z = row[d];
      ztl[(size_t)e * 16 + d] = z;
      ss += z * z;
    }
    for (int q = 0; q < 8; ++q) st[(size_t)e * 8 + q] = row[8 + q];
    c = row[16] + 0.5f * (ss + 16.f * LOG2PI);
  }
  float tot = block_reduce_sum(c);
  if (threadIdx.x == 0) atomicAdd(klz_acc, tot);
}

__global__ void klw_kernel(const float* __restrict__ mu, const float* __restrict__ lv,
                           float* __restrict__ acc) {
  int i = blockIdx.x * blockDim.x + threadIdx.x;
  float c = 0.f;
  if (i < 3808) {
    float m = mu[i], l = lv[i];
    c = 0.5f * (m * m + expf(l) - l - 1.f);
  }
  float t = block_reduce_sum(c);
  if (threadIdx.x == 0) atomicAdd(acc, t);
}

// ---------------- fc3 -> NHWC layout: sb[img][pix(4x4)][ci(32)] ----------------
__global__ void fc3_kernel(const float* __restrict__ st, const float* __restrict__ w,
                           const float* __restrict__ b, float* __restrict__ s,
                           int imgBase, int nImg) {
  int e = blockIdx.x * blockDim.x + threadIdx.x;
  if (e >= nImg * 512) return;
  int il = e >> 9; int o = e & 511;          // o = ci*16 + pix (reference reshape)
  const float* sr = st + (size_t)(imgBase + il) * 8;
  float acc = b[o];
#pragma unroll
  for (int q = 0; q < 8; ++q) acc += sr[q] * w[q * 512 + o];
  int ci = o >> 4, pix = o & 15;
  s[((size_t)il * 16 + pix) * 32 + ci] = acc;
}

// ---------------- transposed conv, implicit-GEMM WMMA, NHWC in/out -----------
// y[img,oy,ox,co] = relu(b[co] + sum_{ky,kx,ci} w[K-1-ky,K-1-kx,ci,co]*x[img,iy,ix,ci])
// iy=(oy+ky-lo)/stride when divisible & in range. Cin must be a multiple of 32.
__global__ __launch_bounds__(128)
void convt_wmma_kernel(const float* __restrict__ x, const float* __restrict__ w,
                       const float* __restrict__ bias, float* __restrict__ y,
                       int npix, int Cin, int Cout, int Hin, int Win,
                       int Hout, int Wout, int Kk, int stride, int lo) {
  __shared__ bf16_t Bt[16][LDP];        // Bt[n(co)][k(ci)]
  __shared__ bf16_t At[4][2][16][LDP];  // per-wave, 2 sub-tiles
  const int tid = threadIdx.x;
  const int lane = tid & 31;
  const int wv = tid >> 5;
  const int mbase = (blockIdx.x * 8 + wv * 2) * 16;   // 2 M-tiles per wave
  const int cobase = blockIdx.y * 16;
  const int HWout = Hout * Wout;
  const int kchunks = Cin >> 5;

  v8f acc0 = {};
  v8f acc1 = {};

  // lane covers A row m=lane/2, k-half akb (same for both sub-tiles)
  const int am = lane >> 1;
  const int akb = (lane & 1) << 4;
  int aimg[2], aoy[2], aox[2];
  bool apix[2];
#pragma unroll
  for (int s = 0; s < 2; ++s) {
    int ap = mbase + s * 16 + am;
    apix[s] = (ap < npix);
    aimg[s] = 0; aoy[s] = 0; aox[s] = 0;
    if (apix[s]) {
      aimg[s] = ap / HWout; int rem = ap - aimg[s] * HWout;
      aoy[s] = rem / Wout; aox[s] = rem - aoy[s] * Wout;
    }
  }
  // B staging assignment (uniform, branch-free): thread -> (nn, k0) and (nn, k0+16)
  const int bnn = tid & 15;
  const int bk0 = (tid >> 4) << 1;   // 0..14, even

  for (int ky = 0; ky < Kk; ++ky) {
    for (int kx = 0; kx < Kk; ++kx) {
      bool ok[2]; int iy[2], ix[2];
#pragma unroll
      for (int s = 0; s < 2; ++s) {
        int ry = aoy[s] + ky - lo;
        int rx = aox[s] + kx - lo;
        bool o1 = apix[s] && ry >= 0 && (ry % stride) == 0 && (ry / stride) < Hin;
        bool o2 = rx >= 0 && (rx % stride) == 0 && (rx / stride) < Win;
        ok[s] = o1 && o2;
        iy[s] = ok[s] ? (ry / stride) : 0;
        ix[s] = ok[s] ? (rx / stride) : 0;
      }
      const int ftap = (Kk - 1 - ky) * Kk + (Kk - 1 - kx);
      const float* wb = w + (size_t)ftap * Cin * Cout + cobase;
      for (int kc = 0; kc < kchunks; ++kc) {
        // ---- stage B (straight-line, 2 packed pairs per thread) ----
        {
          const float* wp0 = wb + (size_t)((kc << 5) + bk0) * Cout + bnn;
          const float* wp1 = wp0 + (size_t)16 * Cout;
          *(unsigned*)&Bt[bnn][bk0]      = pack2bf(wp0[0], wp0[Cout]);
          *(unsigned*)&Bt[bnn][bk0 + 16] = pack2bf(wp1[0], wp1[Cout]);
        }
        // ---- stage A (per wave, NHWC: 16 contiguous floats per lane) ----
#pragma unroll
        for (int s = 0; s < 2; ++s) {
          float4 q0 = {0.f, 0.f, 0.f, 0.f}, q1 = q0, q2 = q0, q3 = q0;
          if (ok[s]) {
            const float4* xp = (const float4*)(x +
                ((size_t)aimg[s] * Hin * Win + (size_t)iy[s] * Win + ix[s]) * Cin +
                (kc << 5) + akb);
            q0 = xp[0]; q1 = xp[1]; q2 = xp[2]; q3 = xp[3];
          }
          unsigned* dst = (unsigned*)&At[wv][s][am][akb];
          dst[0] = pack2bf(q0.x, q0.y);
          dst[1] = pack2bf(q0.z, q0.w);
          dst[2] = pack2bf(q1.x, q1.y);
          dst[3] = pack2bf(q1.z, q1.w);
          dst[4] = pack2bf(q2.x, q2.y);
          dst[5] = pack2bf(q2.z, q2.w);
          dst[6] = pack2bf(q3.x, q3.y);
          dst[7] = pack2bf(q3.z, q3.w);
        }
        __syncthreads();
        // ---- fragments (ISA 16-bit A/B layouts) ----
        v16bf bfr;
        const bf16_t* br = &Bt[lane & 15][0];
        const int kb0 = (lane < 16) ? 0 : 8;
#pragma unroll
        for (int j = 0; j < 8; ++j) { bfr[j] = br[kb0 + j]; bfr[8 + j] = br[kb0 + 16 + j]; }
        const int ka0 = (lane < 16) ? 0 : 8;
        {
          v16bf af;
          const bf16_t* ar = &At[wv][0][lane & 15][0];
#pragma unroll
          for (int j = 0; j < 8; ++j) { af[j] = ar[ka0 + j]; af[8 + j] = ar[ka0 + 16 + j]; }
          acc0 = __builtin_amdgcn_wmma_f32_16x16x32_bf16(false, af, false, bfr,
                                                         (short)0, acc0, false, false);
        }
        {
          v16bf af;
          const bf16_t* ar = &At[wv][1][lane & 15][0];
#pragma unroll
          for (int j = 0; j < 8; ++j) { af[j] = ar[ka0 + j]; af[8 + j] = ar[ka0 + 16 + j]; }
          acc1 = __builtin_amdgcn_wmma_f32_16x16x32_bf16(false, af, false, bfr,
                                                         (short)0, acc1, false, false);
        }
        __syncthreads();
      }
    }
  }
  // ---- store D (NHWC: co contiguous across lanes) ----
  const int col = cobase + (lane & 15);
  const float bv = bias[col];
#pragma unroll
  for (int s = 0; s < 2; ++s) {
    const v8f* ac = (s == 0) ? &acc0 : &acc1;
#pragma unroll
    for (int r = 0; r < 8; ++r) {
      int mrow = (lane < 16) ? r : (8 + r);
      int p = mbase + s * 16 + mrow;
      if (p < npix) {
        float v = fmaxf((*ac)[r] + bv, 0.f);
        y[(size_t)p * Cout + col] = v;
      }
    }
  }
}

// ---------------- final convT (Cout=1) + sigmoid, NHWC input ----------------
__global__ void convt4_kernel(const float* __restrict__ x, const float* __restrict__ w,
                              const float* __restrict__ b, float* __restrict__ xrec,
                              int imgBase, int nImg) {
  int e = blockIdx.x * blockDim.x + threadIdx.x;
  if (e >= nImg * 784) return;
  int il = e / 784; int pix = e - il * 784;
  int oy = pix / 28, ox = pix - oy * 28;
  float acc = b[0];
  for (int ky = 0; ky < 5; ++ky) {
    int iy = oy + ky - 2; if (iy < 0 || iy >= 28) continue;
    for (int kx = 0; kx < 5; ++kx) {
      int ix = ox + kx - 2; if (ix < 0 || ix >= 28) continue;
      const float* wp = w + ((4 - ky) * 5 + (4 - kx)) * 16;
      const float4* xp = (const float4*)(x + ((size_t)il * 784 + iy * 28 + ix) * 16);
      float4 x0 = xp[0], x1 = xp[1], x2 = xp[2], x3 = xp[3];
      acc += x0.x * wp[0] + x0.y * wp[1] + x0.z * wp[2] + x0.w * wp[3];
      acc += x1.x * wp[4] + x1.y * wp[5] + x1.z * wp[6] + x1.w * wp[7];
      acc += x2.x * wp[8] + x2.y * wp[9] + x2.z * wp[10] + x2.w * wp[11];
      acc += x3.x * wp[12] + x3.y * wp[13] + x3.z * wp[14] + x3.w * wp[15];
    }
  }
  xrec[(size_t)(imgBase + il) * 784 + pix] = 1.f / (1.f + expf(-acc));
}

__global__ void lhood_kernel(const float* __restrict__ X, const float* __restrict__ xrec,
                             int elemBase, int total, float* __restrict__ acc) {
  int i = blockIdx.x * blockDim.x + threadIdx.x;
  float c = 0.f;
  if (i < total) {
    size_t g = (size_t)elemBase + i;
    float p = xrec[g];
    p = fminf(fmaxf(p, 1e-6f), 1.f - 1e-6f);
    float xv = X[g];
    c = logf(p) * xv + log1pf(-p) * (1.f - xv);
  }
  float t = block_reduce_sum(c);
  if (threadIdx.x == 0) atomicAdd(acc, t);
}

__global__ void finalize_kernel(const float* __restrict__ acc, const int* __restrict__ ndata,
                                float* __restrict__ out_scal) {
  float lhood = acc[0] / 512.f;
  float klz = acc[1] / 512.f;
  float klw = acc[2];
  float Nd = (float)ndata[0];
  out_scal[0] = Nd * (lhood - klz) - klw;  // elbo
  out_scal[1] = lhood;
  out_scal[2] = klz;
  out_scal[3] = klw;
}

// ---------------- host orchestration ----------------
static const double DP_A2[6] = {1.0 / 5, 0, 0, 0, 0, 0};
static const double DP_A3[6] = {3.0 / 40, 9.0 / 40, 0, 0, 0, 0};
static const double DP_A4[6] = {44.0 / 45, -56.0 / 15, 32.0 / 9, 0, 0, 0};
static const double DP_A5[6] = {19372.0 / 6561, -25360.0 / 2187, 64448.0 / 6561, -212.0 / 729, 0, 0};
static const double DP_A6[6] = {9017.0 / 3168, -355.0 / 33, 46732.0 / 5247, 49.0 / 176, -5103.0 / 18656, 0};
static const double DP_B[6]  = {35.0 / 384, 0, 500.0 / 1113, 125.0 / 192, -2187.0 / 6784, 11.0 / 84};

extern "C" void kernel_launch(void* const* d_in, const int* in_sizes, int n_in,
                              void* d_out, int out_size, void* d_ws, size_t ws_size,
                              hipStream_t stream) {
  (void)in_sizes; (void)n_in; (void)out_size; (void)ws_size;
  const float* X      = (const float*)d_in[0];
  const float* eps    = (const float*)d_in[1];
  const float* enc_w1 = (const float*)d_in[2];
  const float* enc_b1 = (const float*)d_in[3];
  const float* enc_w2 = (const float*)d_in[4];
  const float* enc_b2 = (const float*)d_in[5];
  const float* enc_w3 = (const float*)d_in[6];
  const float* enc_b3 = (const float*)d_in[7];
  const float* fc1_w  = (const float*)d_in[8];
  const float* fc1_b  = (const float*)d_in[9];
  const float* fc2_w  = (const float*)d_in[10];
  const float* fc2_b  = (const float*)d_in[11];
  const float* fc3_w  = (const float*)d_in[12];
  const float* fc3_b  = (const float*)d_in[13];
  const float* bW1    = (const float*)d_in[14];
  const float* bB1    = (const float*)d_in[15];
  const float* bW2    = (const float*)d_in[16];
  const float* bB2    = (const float*)d_in[17];
  const float* bW3    = (const float*)d_in[18];
  const float* bB3    = (const float*)d_in[19];
  const float* w_mu   = (const float*)d_in[20];
  const float* w_logv = (const float*)d_in[21];
  const float* dec_w1 = (const float*)d_in[22];
  const float* dec_b1 = (const float*)d_in[23];
  const float* dec_w2 = (const float*)d_in[24];
  const float* dec_b2 = (const float*)d_in[25];
  const float* dec_w3 = (const float*)d_in[26];
  const float* dec_b3 = (const float*)d_in[27];
  const float* dec_w4 = (const float*)d_in[28];
  const float* dec_b4 = (const float*)d_in[29];
  const int*   ndata  = (const int*)d_in[30];
  float* out = (float*)d_out;

  float* W = (float*)d_ws;
  size_t off = 0;
  auto alloc = [&](size_t n) { float* p = W + off; off += n; return p; };
  float* ACC  = alloc(16);
  float* E1   = alloc((size_t)512 * 8 * 14 * 14);
  float* E2   = alloc((size_t)512 * 16 * 7 * 7);
  float* H0   = alloc((size_t)512 * 512);
  float* TRAJ = alloc((size_t)16 * N_B * SD);
  float* YI   = alloc((size_t)N_B * SD);
  float* KS   = alloc((size_t)6 * N_B * SD);
  float* L1   = alloc((size_t)512 * 50);
  float* DC1  = alloc((size_t)512 * 50);
  float* L2   = alloc((size_t)512 * 50);
  float* DC2  = alloc((size_t)512 * 50);
  float* R1p  = alloc(512);
  float* R2p  = alloc(512);
  float* DV   = alloc((size_t)512 * 8);
  float* TOUT = alloc((size_t)8 * 512);
  float* ST   = alloc((size_t)8192 * 8);
  const int CHUNK = 1024;
  float* SB   = alloc((size_t)CHUNK * 512);
  float* D1b  = alloc((size_t)CHUNK * 64 * 36);
  float* D2b  = alloc((size_t)CHUNK * 32 * 169);
  float* D3b  = alloc((size_t)CHUNK * 16 * 784);

  const size_t OFF_QM   = 6422528;               // after Xrec (512*16*784)
  const size_t OFF_QV   = OFF_QM + 8192;
  const size_t OFF_ZTL  = OFF_QV + 8192;
  const size_t OFF_SCAL = OFF_ZTL + 131072;

  zero_kernel<<<1, 32, 0, stream>>>(ACC, 16);

  // ---- encoder on X[:,0]
  {
    int tot = 512 * 8 * 14 * 14;
    conv_relu_kernel<<<(tot + 255) / 256, 256, 0, stream>>>(
        X, enc_w1, enc_b1, E1, 512, 1, 28, 28, 8, 14, 14, 5, 2, 2, T_N * 784);
    tot = 512 * 16 * 7 * 7;
    conv_relu_kernel<<<(tot + 255) / 256, 256, 0, stream>>>(
        E1, enc_w2, enc_b2, E2, 512, 8, 14, 14, 16, 7, 7, 5, 2, 2, 8 * 14 * 14);
    tot = 512 * 32 * 4 * 4;
    conv_relu_kernel<<<(tot + 255) / 256, 256, 0, stream>>>(
        E2, enc_w3, enc_b3, H0, 512, 16, 7, 7, 32, 4, 4, 5, 2, 2, 16 * 7 * 7);
  }
  fc_kernel<<<(8192 + 255) / 256, 256, 0, stream>>>(H0, fc1_w, fc1_b, out + OFF_QM, 512, 512, 16);
  fc_kernel<<<(8192 + 255) / 256, 256, 0, stream>>>(H0, fc2_w, fc2_b, out + OFF_QV, 512, 512, 16);
  init_y0_kernel<<<2, 256, 0, stream>>>(out + OFF_QM, out + OFF_QV, eps, TRAJ);

  // ---- dopri5 fixed-step integration
  auto rhs = [&](const float* ys, float* kout) {
    bnn_forward_kernel<<<512, 32, 0, stream>>>(ys, bW1, bB1, bW2, bB2, bW3, bB3,
                                               L1, DC1, R1p, L2, DC2, R2p, DV);
    bnn_tangent_kernel<<<4096, 32, 0, stream>>>(bW1, bW2, bW3, L1, DC1, R1p, L2, DC2, R2p, TOUT);
    assemble_k_kernel<<<2, 256, 0, stream>>>(ys, DV, TOUT, kout);
  };
  auto comb = [&](const float* ybase, const double* c, float* dst) {
    int tot = N_B * SD;
    ode_combine_kernel<<<(tot + 255) / 256, 256, 0, stream>>>(
        ybase, KS, KS + N_B * SD, KS + 2 * N_B * SD, KS + 3 * N_B * SD,
        KS + 4 * N_B * SD, KS + 5 * N_B * SD,
        (float)c[0], (float)c[1], (float)c[2], (float)c[3], (float)c[4], (float)c[5],
        0.1f, dst);
  };
  for (int t = 0; t < T_N - 1; ++t) {
    float* ycur = TRAJ + (size_t)t * N_B * SD;
    float* ynext = TRAJ + (size_t)(t + 1) * N_B * SD;
    rhs(ycur, KS);
    comb(ycur, DP_A2, YI); rhs(YI, KS + 1 * N_B * SD);
    comb(ycur, DP_A3, YI); rhs(YI, KS + 2 * N_B * SD);
    comb(ycur, DP_A4, YI); rhs(YI, KS + 3 * N_B * SD);
    comb(ycur, DP_A5, YI); rhs(YI, KS + 4 * N_B * SD);
    comb(ycur, DP_A6, YI); rhs(YI, KS + 5 * N_B * SD);
    comb(ycur, DP_B, ynext);
  }

  // ---- ztL output + st extraction + kl_z, kl_w
  ztl_klz_kernel<<<32, 256, 0, stream>>>(TRAJ, out + OFF_ZTL, ST, ACC + 1);
  klw_kernel<<<(3808 + 255) / 256, 256, 0, stream>>>(w_mu, w_logv, ACC + 2);

  // ---- decoder, 8 chunks of 1024 images (L2 resident), NHWC internally
  auto convt = [&](const float* xin, const float* wgt, const float* bias, float* yo,
                   int Cin, int Cout, int Hin, int Win, int Hout, int Wout,
                   int Kk, int stride, int lo) {
    int npix = CHUNK * Hout * Wout;
    int mtiles = (npix + 15) / 16;
    dim3 grid((mtiles + 7) / 8, Cout / 16);
    convt_wmma_kernel<<<grid, 128, 0, stream>>>(xin, wgt, bias, yo, npix, Cin, Cout,
                                                Hin, Win, Hout, Wout, Kk, stride, lo);
  };
  for (int cb = 0; cb < 8192; cb += CHUNK) {
    fc3_kernel<<<(CHUNK * 512 + 255) / 256, 256, 0, stream>>>(ST, fc3_w, fc3_b, SB, cb, CHUNK);
    convt(SB,  dec_w1, dec_b1, D1b, 32, 64, 4, 4, 6, 6, 3, 1, 2);
    convt(D1b, dec_w2, dec_b2, D2b, 64, 32, 6, 6, 13, 13, 5, 2, 3);
    convt(D2b, dec_w3, dec_b3, D3b, 32, 16, 13, 13, 28, 28, 5, 2, 3);
    int tot = CHUNK * 784;
    convt4_kernel<<<(tot + 255) / 256, 256, 0, stream>>>(D3b, dec_w4, dec_b4, out, cb, CHUNK);
    lhood_kernel<<<(tot + 255) / 256, 256, 0, stream>>>(X, out, cb * 784, tot, ACC);
  }
  finalize_kernel<<<1, 1, 0, stream>>>(ACC, ndata, out + OFF_SCAL);
}
